// SymmetricCrossAttentionBlock_824633721038
// MI455X (gfx1250) — matrix operand order repaired and checked by reading, not compiled, not run on hardware
//
#include <hip/hip_runtime.h>

// CDNA5 / gfx1250 implementation of the symmetric cross-attention block.
// All GEMMs run on v_wmma_f32_16x16x32_f16 (f16 in, f32 accumulate).
// Key algebraic simplifications vs the reference:
//  * dec_cross has a single key/value token -> softmax==1 -> collapses to
//    dcv[j,b] = Wo_dc(Wv_dc f_n[j,b] + bv) + bo  (G_n = LN(dec,n_dc) unused)
//  * pooling weights sum to 1 -> dec1 = dec + sum_j w_j * dcv_j
// Big intermediates (G_im, K/V) kept in f16 so they live in the 192MB L2.
// Wave tile 32M x 64N: 8 wmma per 32-K step vs 12 b128 loads.

typedef _Float16 h16;
typedef __attribute__((ext_vector_type(16))) _Float16 v16h;
typedef __attribute__((ext_vector_type(8)))  _Float16 v8h;
typedef __attribute__((ext_vector_type(8)))  float    v8f;

#define D512 512

// ---------------------------------------------------------------- cast f32->f16
__global__ void cast_kernel(const float* __restrict__ x, h16* __restrict__ y, int n) {
    int t = blockIdx.x * blockDim.x + threadIdx.x;
    if (t < n) y[t] = (h16)x[t];
}

// ---------------------------------------------------------------- LayerNorm rows of 512
// mode 0: src row = X + row*512
// mode 1: G_fj row (j,i,b): src = dec[(i,b)] + dcv[(j,b)]   (row = ((j*64+i)*16+b))
__global__ void ln_kernel(const float* __restrict__ X, const float* __restrict__ X2,
                          const float* __restrict__ g, const float* __restrict__ bb,
                          h16* __restrict__ outH, int mode) {
    int row = blockIdx.x;
    const float* s1;
    const float* s2 = nullptr;
    if (mode == 0) {
        s1 = X + (size_t)row * D512;
    } else {
        int b = row & 15, i = (row >> 4) & 63, j = row >> 10;
        s1 = X  + (size_t)(i * 16 + b) * D512;
        s2 = X2 + (size_t)(j * 16 + b) * D512;
    }
    int t = threadIdx.x;                    // 128 threads, 4 elems each
    float v[4], sum = 0.f, sq = 0.f;
    #pragma unroll
    for (int u = 0; u < 4; ++u) {
        int c = t + u * 128;
        float x = s1[c];
        if (s2) x += s2[c];
        v[u] = x; sum += x; sq += x * x;
    }
    #pragma unroll
    for (int o = 16; o > 0; o >>= 1) { sum += __shfl_down(sum, o, 32); sq += __shfl_down(sq, o, 32); }
    __shared__ float sh[8];
    if ((t & 31) == 0) { sh[t >> 5] = sum; sh[4 + (t >> 5)] = sq; }
    __syncthreads();
    float S = sh[0] + sh[1] + sh[2] + sh[3];
    float Q = sh[4] + sh[5] + sh[6] + sh[7];
    float mean = S * (1.f / 512.f);
    float var  = Q * (1.f / 512.f) - mean * mean;
    float rstd = rsqrtf(var + 1e-5f);
    #pragma unroll
    for (int u = 0; u < 4; ++u) {
        int c = t + u * 128;
        outH[(size_t)row * D512 + c] = (h16)((v[u] - mean) * rstd * g[c] + bb[c]);
    }
}

// ---------------------------------------------------------------- generic WMMA GEMM
// y[M,N] = A[M,K] @ W[N,K]^T + bias[N]  (+ resid[M,N]) (optional ReLU)
// Block = 256 thr = 8 waves (2 M-waves x 4 N-waves); block tile 64M x 256N;
// wave tile 32M x 64N -> acc[2][4], 8 wmma per K-step.
__global__ void gemm_kernel(const h16* __restrict__ A, const h16* __restrict__ W,
                            const float* __restrict__ bias, const float* __restrict__ resid,
                            float* __restrict__ outF, h16* __restrict__ outH,
                            int M, int N, int K, int relu) {
    const int lane = threadIdx.x & 31;
    const int wave = threadIdx.x >> 5;
    const int wn = wave & 3, wm = wave >> 2;
    const int m0 = blockIdx.x * 64 + wm * 32;
    const int n0 = blockIdx.y * 256 + wn * 64;
    const int lm = lane & 15, lh = lane >> 4;

    v8f acc[2][4] = {};
    const h16* arow0 = A + (size_t)(m0 + lm) * K;
    const h16* arow1 = A + (size_t)(m0 + 16 + lm) * K;
    for (int k0 = 0; k0 < K; k0 += 32) {
        if (k0 + 128 < K) {                              // pull next A chunks toward WGP
            __builtin_prefetch(arow0 + k0 + 128, 0, 1);  // -> global_prefetch_b8
            __builtin_prefetch(arow1 + k0 + 128, 0, 1);
        }
        // A fragments: lane(lm,lh) holds A[row, k0+lh*8..+7] and A[row, k0+16+lh*8..+7]
        v8h a00 = *(const v8h*)(arow0 + k0 + lh * 8);
        v8h a01 = *(const v8h*)(arow0 + k0 + 16 + lh * 8);
        v8h a10 = *(const v8h*)(arow1 + k0 + lh * 8);
        v8h a11 = *(const v8h*)(arow1 + k0 + 16 + lh * 8);
        v16h af[2];
        #pragma unroll
        for (int t = 0; t < 8; ++t) {
            af[0][t] = a00[t]; af[0][8 + t] = a01[t];
            af[1][t] = a10[t]; af[1][8 + t] = a11[t];
        }
        #pragma unroll
        for (int s = 0; s < 4; ++s) {
            // B fragment: lane(lm,lh) holds W[n0+s*16+lm, k0+lh*8..+7] and W[.., k0+16+lh*8..+7]
            const h16* wrow = W + (size_t)(n0 + s * 16 + lm) * K + k0;
            v8h b0 = *(const v8h*)(wrow + lh * 8);
            v8h b1 = *(const v8h*)(wrow + 16 + lh * 8);
            v16h bf = __builtin_shufflevector(b0, b1, 0,1,2,3,4,5,6,7,8,9,10,11,12,13,14,15);
            acc[0][s] = __builtin_amdgcn_wmma_f32_16x16x32_f16(
                false, af[0], false, bf, (short)0, acc[0][s], false, false);
            acc[1][s] = __builtin_amdgcn_wmma_f32_16x16x32_f16(
                false, af[1], false, bf, (short)0, acc[1][s], false, false);
        }
    }
    // C/D layout: col n = n0+s*16+lm ; row m = m0 + u*16 + v + 8*lh
    #pragma unroll
    for (int u = 0; u < 2; ++u) {
        #pragma unroll
        for (int s = 0; s < 4; ++s) {
            int n = n0 + s * 16 + lm;
            float bv = bias ? bias[n] : 0.f;
            #pragma unroll
            for (int v = 0; v < 8; ++v) {
                int m = m0 + u * 16 + v + 8 * lh;
                float y = acc[u][s][v] + bv;
                if (resid) y += resid[(size_t)m * N + n];
                if (relu) y = y > 0.f ? y : 0.f;
                size_t idx = (size_t)m * N + n;
                if (outF) outF[idx] = y;
                if (outH) outH[idx] = (h16)y;
            }
        }
    }
}

// ---------------------------------------------------------------- fused pooling-energy GEMM
// e[row] = sum_n tanh( (dec+dcv)[row,:] @ pW^T + pWb + pb )[n] * pv[n]
// A built on the fly: row=((j*64+i)*16+b) -> dech[(i,b)] + dcvh[(j,b)]
__global__ void energy_kernel(const h16* __restrict__ dech, const h16* __restrict__ dcvh,
                              const h16* __restrict__ pWh, const float* __restrict__ pWb,
                              const float* __restrict__ pb, const float* __restrict__ pv,
                              float* __restrict__ e) {
    const int lane = threadIdx.x & 31;
    const int wave = threadIdx.x >> 5;
    const int wn = wave & 3, wm = wave >> 2;
    const int m0 = blockIdx.x * 64 + wm * 32;
    const int n0 = blockIdx.y * 256 + wn * 64;
    const int lm = lane & 15, lh = lane >> 4;
    const int K = D512;

    int mr0 = m0 + lm, mr1 = m0 + 16 + lm;
    const h16* adec0 = dech + (size_t)(mr0 & 1023) * K;
    const h16* adcv0 = dcvh + (size_t)(((mr0 >> 10) << 4) + (mr0 & 15)) * K;
    const h16* adec1 = dech + (size_t)(mr1 & 1023) * K;
    const h16* adcv1 = dcvh + (size_t)(((mr1 >> 10) << 4) + (mr1 & 15)) * K;

    v8f acc[2][4] = {};
    for (int k0 = 0; k0 < K; k0 += 32) {
        v8h x00 = *(const v8h*)(adec0 + k0 + lh * 8);
        v8h y00 = *(const v8h*)(adcv0 + k0 + lh * 8);
        v8h x01 = *(const v8h*)(adec0 + k0 + 16 + lh * 8);
        v8h y01 = *(const v8h*)(adcv0 + k0 + 16 + lh * 8);
        v8h x10 = *(const v8h*)(adec1 + k0 + lh * 8);
        v8h y10 = *(const v8h*)(adcv1 + k0 + lh * 8);
        v8h x11 = *(const v8h*)(adec1 + k0 + 16 + lh * 8);
        v8h y11 = *(const v8h*)(adcv1 + k0 + 16 + lh * 8);
        v16h af[2];
        #pragma unroll
        for (int t = 0; t < 8; ++t) {
            af[0][t] = (h16)(x00[t] + y00[t]); af[0][8 + t] = (h16)(x01[t] + y01[t]);
            af[1][t] = (h16)(x10[t] + y10[t]); af[1][8 + t] = (h16)(x11[t] + y11[t]);
        }
        #pragma unroll
        for (int s = 0; s < 4; ++s) {
            const h16* wrow = pWh + (size_t)(n0 + s * 16 + lm) * K + k0;
            v8h b0 = *(const v8h*)(wrow + lh * 8);
            v8h b1 = *(const v8h*)(wrow + 16 + lh * 8);
            v16h bf = __builtin_shufflevector(b0, b1, 0,1,2,3,4,5,6,7,8,9,10,11,12,13,14,15);
            acc[0][s] = __builtin_amdgcn_wmma_f32_16x16x32_f16(
                false, af[0], false, bf, (short)0, acc[0][s], false, false);
            acc[1][s] = __builtin_amdgcn_wmma_f32_16x16x32_f16(
                false, af[1], false, bf, (short)0, acc[1][s], false, false);
        }
    }
    float biasv[4], pvv[4];
    #pragma unroll
    for (int s = 0; s < 4; ++s) { int n = n0 + s * 16 + lm; biasv[s] = pWb[n] + pb[n]; pvv[s] = pv[n]; }
    #pragma unroll
    for (int u = 0; u < 2; ++u) {
        #pragma unroll
        for (int v = 0; v < 8; ++v) {
            float part = 0.f;
            #pragma unroll
            for (int s = 0; s < 4; ++s) part += tanhf(acc[u][s][v] + biasv[s]) * pvv[s];
            #pragma unroll
            for (int o = 1; o < 16; o <<= 1) part += __shfl_xor(part, o, 32);  // reduce 16 n-lanes
            if (lm == 0) atomicAdd(e + (m0 + u * 16 + v + 8 * lh), part);
        }
    }
}

// ---------------------------------------------------------------- pooling softmax over j (in place)
__global__ void pool_softmax(float* __restrict__ e) {
    int t = blockIdx.x * blockDim.x + threadIdx.x;   // t = i*16+b in [0,1024)
    if (t >= 1024) return;
    float mx = -1e30f;
    for (int j = 0; j < 64; ++j) mx = fmaxf(mx, e[t + j * 1024]);
    float s = 0.f;
    for (int j = 0; j < 64; ++j) { float w = __expf(e[t + j * 1024] - mx); e[t + j * 1024] = w; s += w; }
    float inv = 1.f / s;
    for (int j = 0; j < 64; ++j) e[t + j * 1024] *= inv;
}

// dec1[i,b,:] = dec[i,b,:] + sum_j w[j,i,b] * dcv[j,b,:]
__global__ void dec1_kernel(const float* __restrict__ dec, const float* __restrict__ dcvf,
                            const float* __restrict__ w, float* __restrict__ out) {
    int t = blockIdx.x * blockDim.x + threadIdx.x;   // 1024*512 threads
    int c = t & 511, m = t >> 9;                     // m = i*16+b
    int i = m >> 4, b = m & 15;
    float acc = dec[(size_t)m * D512 + c];
    for (int j = 0; j < 64; ++j)
        acc += w[(size_t)j * 1024 + i * 16 + b] * dcvf[((size_t)(j * 16 + b)) * D512 + c];
    out[(size_t)m * D512 + c] = acc;
}

// ---------------------------------------------------------------- enc_cross attention (len-64 keys)
__global__ void ec_scores(const h16* __restrict__ q, const h16* __restrict__ kh,
                          float* __restrict__ s) {
    int blk = blockIdx.x; int j = blk >> 4, b = blk & 15;
    int t = threadIdx.x;  int h = t >> 6, i = t & 63;
    const h16* qp = q  + ((size_t)(j * 16 + b)) * D512 + h * 64;
    const h16* kp = kh + ((size_t)((j * 64 + i) * 16 + b)) * D512 + h * 64;
    float acc = 0.f;
    for (int e = 0; e < 64; ++e) acc += (float)qp[e] * (float)kp[e];
    s[(((size_t)(j * 16 + b)) * 8 + h) * 64 + i] = acc * 0.125f;   // hd^-0.5
}

__global__ void softmax64(float* __restrict__ s, int rows) {
    int r = blockIdx.x * blockDim.x + threadIdx.x;
    if (r >= rows) return;
    float* p = s + (size_t)r * 64;
    float mx = -1e30f;
    for (int i = 0; i < 64; ++i) mx = fmaxf(mx, p[i]);
    float sum = 0.f;
    for (int i = 0; i < 64; ++i) { float w = __expf(p[i] - mx); p[i] = w; sum += w; }
    float inv = 1.f / sum;
    for (int i = 0; i < 64; ++i) p[i] *= inv;
}

__global__ void ec_attout(const float* __restrict__ w, const h16* __restrict__ vh,
                          h16* __restrict__ o) {
    int blk = blockIdx.x; int j = blk >> 4, b = blk & 15;
    int t = threadIdx.x;  int h = t >> 6, e = t & 63;
    const float* wr = w + (((size_t)(j * 16 + b)) * 8 + h) * 64;
    float acc = 0.f;
    for (int i = 0; i < 64; ++i)
        acc += wr[i] * (float)vh[((size_t)((j * 64 + i) * 16 + b)) * D512 + h * 64 + e];
    o[((size_t)(j * 16 + b)) * D512 + h * 64 + e] = (h16)acc;
}

// ---------------------------------------------------------------- self-attention (L=64)
__global__ void sa_scores(const h16* __restrict__ qkv, float* __restrict__ s) {
    int t = blockIdx.x * blockDim.x + threadIdx.x;   // 524288
    int ki = t & 63, qi = (t >> 6) & 63, h = (t >> 12) & 7, b = t >> 15;
    const h16* qp = qkv + ((size_t)(qi * 16 + b)) * 1536 + h * 64;
    const h16* kp = qkv + ((size_t)(ki * 16 + b)) * 1536 + 512 + h * 64;
    float acc = 0.f;
    for (int e = 0; e < 64; ++e) acc += (float)qp[e] * (float)kp[e];
    s[(((size_t)(b * 8 + h)) * 64 + qi) * 64 + ki] = acc * 0.125f;
}

__global__ void sa_out(const h16* __restrict__ qkv, const float* __restrict__ w,
                       h16* __restrict__ o) {
    int t = blockIdx.x * blockDim.x + threadIdx.x;   // 524288
    int e = t & 63, h = (t >> 6) & 7, b = (t >> 9) & 15, l = t >> 13;
    const float* wr = w + (((size_t)(b * 8 + h)) * 64 + l) * 64;
    float acc = 0.f;
    for (int ki = 0; ki < 64; ++ki)
        acc += wr[ki] * (float)qkv[((size_t)(ki * 16 + b)) * 1536 + 1024 + h * 64 + e];
    o[((size_t)(l * 16 + b)) * D512 + h * 64 + e] = (h16)acc;
}

// ================================================================ host side
extern "C" void kernel_launch(void* const* d_in, const int* in_sizes, int n_in,
                              void* d_out, int out_size, void* d_ws, size_t ws_size,
                              hipStream_t stream) {
    (void)in_sizes; (void)n_in; (void)out_size; (void)ws_size;
    const float* enc   = (const float*)d_in[0];
    const float* dec   = (const float*)d_in[1];

    // workspace carve-up (256B aligned)
    char* wp = (char*)d_ws;
    auto alloc = [&](size_t bytes) -> char* { char* r = wp; wp += (bytes + 255) & ~(size_t)255; return r; };
    auto ah = [&](size_t n) -> h16*   { return (h16*)alloc(n * sizeof(h16)); };
    auto af = [&](size_t n) -> float* { return (float*)alloc(n * sizeof(float)); };

    const size_t R  = 1024;          // NE*B == ND*B rows
    const size_t BIG = 65536;        // ne*nd*B rows

    h16 *whECWI = ah(1536 * 512), *whECWO = ah(512 * 512);
    h16 *whDCWV = ah(512 * 512),  *whDCWO = ah(512 * 512);
    h16 *whESWI = ah(1536 * 512), *whESWO = ah(512 * 512);
    h16 *whDSWI = ah(1536 * 512), *whDSWO = ah(512 * 512);
    h16 *whEF1  = ah(2048 * 512), *whEF2  = ah(512 * 2048);
    h16 *whDF1  = ah(2048 * 512), *whDF2  = ah(512 * 2048);
    h16 *whPW   = ah(512 * 512);
    h16 *dech = ah(R * 512), *fnh = ah(R * 512), *tmph = ah(R * 512);
    h16 *dcvh = ah(R * 512), *qech = ah(R * 512), *oech = ah(R * 512);
    h16 *lnh  = ah(R * 512), *oh = ah(R * 512);
    h16 *qkvh = ah(R * 1536), *hidh = ah(R * 2048);
    h16 *gimh = ah(BIG * 512), *khv = ah(BIG * 512);
    float *dcvf = af(R * 512), *att_s = af(BIG * 8 /* 524288 */), *evec = af(BIG);
    float *enc1f = af(R * 512), *dec1f = af(R * 512), *enc2f = af(R * 512), *dec2f = af(R * 512);

    float* outEnc = (float*)d_out;
    float* outDec = outEnc + R * 512;

    auto cast = [&](const float* src, h16* dst, size_t n) {
        cast_kernel<<<dim3((unsigned)((n + 255) / 256)), 256, 0, stream>>>(src, dst, (int)n);
    };
    auto gemm = [&](const h16* A, const h16* W, const float* bias, const float* resid,
                    float* oF, h16* oH, int M, int N, int K, int relu) {
        gemm_kernel<<<dim3(M / 64, N / 256), 256, 0, stream>>>(A, W, bias, resid, oF, oH, M, N, K, relu);
    };

    // ---- weight casts (f32 -> f16, row-major preserved)
    cast((const float*)d_in[2],  whECWI, 1536 * 512);
    cast((const float*)d_in[4],  whECWO, 512 * 512);
    cast((const float*)d_in[6] + 1024 * 512, whDCWV, 512 * 512);   // Wv_dc rows [2d,3d)
    cast((const float*)d_in[8],  whDCWO, 512 * 512);
    cast((const float*)d_in[10], whESWI, 1536 * 512);
    cast((const float*)d_in[12], whESWO, 512 * 512);
    cast((const float*)d_in[14], whDSWI, 1536 * 512);
    cast((const float*)d_in[16], whDSWO, 512 * 512);
    cast((const float*)d_in[18], whEF1, 2048 * 512);
    cast((const float*)d_in[20], whEF2, 512 * 2048);
    cast((const float*)d_in[22], whDF1, 2048 * 512);
    cast((const float*)d_in[24], whDF2, 512 * 2048);
    cast((const float*)d_in[40], whPW, 512 * 512);
    cast(dec, dech, R * 512);

    // ---- f_n = LN(enc; n_ec)
    ln_kernel<<<1024, 128, 0, stream>>>(enc, nullptr, (const float*)d_in[26], (const float*)d_in[27], fnh, 0);

    // ---- dcv = Wo_dc(Wv_dc f_n + bv) + bo   (collapsed dec_cross)
    gemm(fnh, whDCWV, (const float*)d_in[7] + 1024, nullptr, nullptr, tmph, 1024, 512, 512, 0);
    gemm(tmph, whDCWO, (const float*)d_in[9], nullptr, dcvf, dcvh, 1024, 512, 512, 0);

    // ---- G_im = LN(dec + dcv ; n_dci)  (65536 rows, fused construction)
    ln_kernel<<<65536, 128, 0, stream>>>(dec, dcvf, (const float*)d_in[34], (const float*)d_in[35], gimh, 1);

    // ---- enc_cross: q = Wq_ec f_n + bq ; K/V from G_im ; softmax over 64 dec tokens
    gemm(fnh, whECWI, (const float*)d_in[3], nullptr, nullptr, qech, 1024, 512, 512, 0);
    gemm(gimh, whECWI + (size_t)512 * 512, (const float*)d_in[3] + 512, nullptr, nullptr, khv, 65536, 512, 512, 0);
    ec_scores<<<1024, 512, 0, stream>>>(qech, khv, att_s);
    softmax64<<<32, 256, 0, stream>>>(att_s, 8192);
    gemm(gimh, whECWI + (size_t)1024 * 512, (const float*)d_in[3] + 1024, nullptr, nullptr, khv, 65536, 512, 512, 0);
    ec_attout<<<1024, 512, 0, stream>>>(att_s, khv, oech);
    gemm(oech, whECWO, (const float*)d_in[5], enc, enc1f, nullptr, 1024, 512, 512, 0);  // enc1 = enc + ec_out

    // ---- attention pooling over ne fusion copies
    hipMemsetAsync(evec, 0, BIG * sizeof(float), stream);
    energy_kernel<<<dim3(1024, 2), 256, 0, stream>>>(dech, dcvh, whPW,
        (const float*)d_in[41], (const float*)d_in[43], (const float*)d_in[42], evec);
    pool_softmax<<<4, 256, 0, stream>>>(evec);
    dec1_kernel<<<2048, 256, 0, stream>>>(dec, dcvf, evec, dec1f);

    // ---- self-attention (pre-norm residual), shared routine for enc/dec
    auto self_attn = [&](const float* xin, const h16* Wih, const float* bi,
                         const h16* Woh, const float* bo,
                         const float* g, const float* bln, float* xout) {
        ln_kernel<<<1024, 128, 0, stream>>>(xin, nullptr, g, bln, lnh, 0);
        gemm(lnh, Wih, bi, nullptr, nullptr, qkvh, 1024, 1536, 512, 0);
        sa_scores<<<2048, 256, 0, stream>>>(qkvh, att_s);
        softmax64<<<32, 256, 0, stream>>>(att_s, 8192);
        sa_out<<<2048, 256, 0, stream>>>(qkvh, att_s, oh);
        gemm(oh, Woh, bo, xin, xout, nullptr, 1024, 512, 512, 0);
    };
    self_attn(enc1f, whESWI, (const float*)d_in[11], whESWO, (const float*)d_in[13],
              (const float*)d_in[28], (const float*)d_in[29], enc2f);
    self_attn(dec1f, whDSWI, (const float*)d_in[15], whDSWO, (const float*)d_in[17],
              (const float*)d_in[36], (const float*)d_in[37], dec2f);

    // ---- FFN (pre-norm residual), writes final outputs
    auto ffn = [&](const float* xin, const h16* W1h, const float* b1,
                   const h16* W2h, const float* b2,
                   const float* g, const float* bln, float* xout) {
        ln_kernel<<<1024, 128, 0, stream>>>(xin, nullptr, g, bln, lnh, 0);
        gemm(lnh, W1h, b1, nullptr, nullptr, hidh, 1024, 2048, 512, 1);     // ReLU
        gemm(hidh, W2h, b2, xin, xout, nullptr, 1024, 512, 2048, 0);
    };
    ffn(enc2f, whEF1, (const float*)d_in[19], whEF2, (const float*)d_in[21],
        (const float*)d_in[30], (const float*)d_in[31], outEnc);
    ffn(dec2f, whDF1, (const float*)d_in[23], whDF2, (const float*)d_in[25],
        (const float*)d_in[38], (const float*)d_in[39], outDec);
}
